// CoefficientLayer_10402410791125
// MI455X (gfx1250) — compile-verified, matrix operand order checked by reading, not run on hardware
//
#include <hip/hip_runtime.h>
#include <hip/hip_bf16.h>
#include <math.h>

// ---------------------------------------------------------------------------
// Types for CDNA5 WMMA (wave32): D(16x16 f32) = A(16x32 bf16) * B(32x16 bf16) + C
// ---------------------------------------------------------------------------
typedef __attribute__((ext_vector_type(16))) __bf16 v16bf;
typedef __attribute__((ext_vector_type(8)))  float  v8f;

#define NATOMS   65536   // 128 * 512
#define NSPEC    4
#define D0       384
#define D1       256
#define D2       192
#define D3       160
#define ALPHA    0.1f
#define INV_ALPHA 10.0f

// workspace layout (bytes)
#define CNT_OFF   0u
#define IDX_OFF   256u
#define W1S_OFF   (IDX_OFF + (unsigned)(NSPEC * NATOMS * 4))                 // 1,048,832
#define W2S_OFF   (W1S_OFF + (unsigned)(NSPEC * 16 * 12 * 1024))            // +786,432
#define W3S_OFF   (W2S_OFF + (unsigned)(NSPEC * 12 * 8  * 1024))            // +393,216
// end = W3S_OFF + NSPEC*10*6*1024 = ~2.36 MB

// ---------------------------------------------------------------------------
// helpers
// ---------------------------------------------------------------------------
__device__ __forceinline__ unsigned short f2bf(float x) {
    union { float f; unsigned u; } c; c.f = x;
    unsigned u = c.u;
    u += 0x7FFFu + ((u >> 16) & 1u);          // round-to-nearest-even
    return (unsigned short)(u >> 16);
}
__device__ __forceinline__ unsigned pack2(float lo, float hi) {
    return (unsigned)f2bf(lo) | ((unsigned)f2bf(hi) << 16);
}

// ---------------------------------------------------------------------------
// kernel 0: zero species counters
// ---------------------------------------------------------------------------
__global__ void zero_counts_kernel(int* counts) {
    if (threadIdx.x < NSPEC) counts[threadIdx.x] = 0;
}

// ---------------------------------------------------------------------------
// kernel 1: bin atoms by species (atomic append into per-species lists)
// ---------------------------------------------------------------------------
__global__ void bin_kernel(const int* __restrict__ species,
                           int* __restrict__ counts,
                           int* __restrict__ idxList) {
    int i = blockIdx.x * blockDim.x + threadIdx.x;
    if (i < NATOMS) {
        int s = species[i];
        int p = atomicAdd(&counts[s], 1);
        idxList[s * NATOMS + p] = i;
    }
}

// ---------------------------------------------------------------------------
// kernel 2: swizzle fp32 weights [S,K,Nc] -> bf16 WMMA B-fragment blocks.
// Block (s, ntile, kk) = 1KB = 256 dwords; dword index r within block:
//   r = h*128 + lane*4 + q  (h = j>>2, q = j&3, j = lane's VGPR 0..7)
//   k = kk*32 + (j>>2)*16 + (lane>>4)*8 + (j&3)*2 ; col = ntile*16 + (lane&15)
// so the GEMM kernel reads each half-fragment as one coalesced b128 per lane.
// ---------------------------------------------------------------------------
__global__ void swizzle_kernel(const float* __restrict__ W, unsigned* __restrict__ out,
                               int K, int Nc, int ntiles, int ksteps) {
    int total = NSPEC * ntiles * ksteps * 256;
    for (int d = blockIdx.x * blockDim.x + threadIdx.x; d < total;
         d += gridDim.x * blockDim.x) {
        int r   = d & 255;
        int blk = d >> 8;
        int kk  = blk % ksteps;
        int t   = blk / ksteps;
        int nt  = t % ntiles;
        int s   = t / ntiles;
        int h = r >> 7, l = (r >> 2) & 31, q = r & 3;
        int j = h * 4 + q;
        int k = kk * 32 + (j >> 2) * 16 + (l >> 4) * 8 + (j & 3) * 2;
        int col = nt * 16 + (l & 15);
        const float* base = W + (size_t)s * K * Nc;
        out[d] = pack2(base[(size_t)k * Nc + col], base[(size_t)(k + 1) * Nc + col]);
    }
}

// ---------------------------------------------------------------------------
// WMMA tile: acc += A(16xK from LDS bf16 row-major) * B(Kx16 from pre-swizzled W)
// ---------------------------------------------------------------------------
template <int KSTEPS, int NTILES>
__device__ __forceinline__ v8f gemm_tile(const unsigned short* sSrc, int ldk,
                                         const unsigned* __restrict__ Wswz,
                                         int s, int wave, int lane) {
    v8f acc = {0.f, 0.f, 0.f, 0.f, 0.f, 0.f, 0.f, 0.f};
    const int m  = lane & 15;
    const int hs = lane >> 4;
#pragma unroll
    for (int kk = 0; kk < KSTEPS; ++kk) {
        union { unsigned u[8]; v16bf h; } A, B;
#pragma unroll
        for (int j = 0; j < 8; ++j) {
            int k = kk * 32 + (j >> 2) * 16 + hs * 8 + (j & 3) * 2;   // k even
            A.u[j] = *(const unsigned*)(sSrc + m * ldk + k);           // ds_load_b32/b128
        }
        const unsigned* p = Wswz + (((unsigned)(s * NTILES + wave) * KSTEPS + kk) << 8);
#pragma unroll
        for (int j = 0; j < 8; ++j)
            B.u[j] = p[(j >> 2) * 128 + lane * 4 + (j & 3)];           // 2x global b128
        acc = __builtin_amdgcn_wmma_f32_16x16x32_bf16(
                  false, A.h, false, B.h, (short)0, acc, false, false);
    }
    return acc;
}

// C-fragment layout: lane holds col = (lane&15); VGPR r -> row r + (lane>>4)*8
__device__ __forceinline__ void epilogue_bf16(v8f acc, const float* __restrict__ bias,
                                              int s, int Ncols, int wave, int lane,
                                              unsigned short* sDst, int ld) {
    int col = wave * 16 + (lane & 15);
    float b = bias[s * Ncols + col];
    int mb = (lane >> 4) * 8;
#pragma unroll
    for (int r = 0; r < 8; ++r) {
        float x = acc[r] + b;
        x = (x > 0.f) ? x : ALPHA * (__expf(x * INV_ALPHA) - 1.f);    // CELU
        sDst[(mb + r) * ld + col] = f2bf(x);
    }
}

__device__ __forceinline__ void epilogue_f32(v8f acc, const float* __restrict__ bias,
                                             int s, int Ncols, int wave, int lane,
                                             float* sDst, int ld) {
    int col = wave * 16 + (lane & 15);
    float b = bias[s * Ncols + col];
    int mb = (lane >> 4) * 8;
#pragma unroll
    for (int r = 0; r < 8; ++r) {
        float x = acc[r] + b;
        x = (x > 0.f) ? x : ALPHA * (__expf(x * INV_ALPHA) - 1.f);
        sDst[(mb + r) * ld + col] = x;
    }
}

// ---------------------------------------------------------------------------
// kernel 3: fused per-species MLP. One block = one 16-atom same-species tile.
// 512 threads = 16 wave32; wave w owns output columns [16w, 16w+16).
// ---------------------------------------------------------------------------
#define LDA  392   // 384 + pad (even)
#define LDH1 264   // 256 + pad
#define LDH2 200   // 192 + pad
#define LDH3 164   // 160 + pad

__global__ __launch_bounds__(512) void
mlp_kernel(const float* __restrict__ aev,
           const float* __restrict__ b1, const float* __restrict__ b2,
           const float* __restrict__ b3, const float* __restrict__ b4,
           const float* __restrict__ W4,
           const float* __restrict__ sb0, const float* __restrict__ sb1,
           const int* __restrict__ counts, const int* __restrict__ idxList,
           const unsigned* __restrict__ W1s, const unsigned* __restrict__ W2s,
           const unsigned* __restrict__ W3s,
           float* __restrict__ out) {
    __shared__ unsigned short sA [16 * LDA];    // 12.25 KB  (aev, bf16)
    __shared__ unsigned short sH1[16 * LDH1];   //  8.25 KB
    __shared__ unsigned short sH2[16 * LDH2];   //  6.25 KB
    __shared__ float          sH3[16 * LDH3];   // 10.25 KB
    __shared__ int atomIdx[16];

    const int tile = blockIdx.x;
    const int s    = blockIdx.y;
    const int cnt  = counts[s];
    if (tile * 16 >= cnt) return;

    const int tid  = threadIdx.x;
    const int wave = tid >> 5;
    const int lane = tid & 31;

    if (tid < 16) {
        int g = tile * 16 + tid;
        atomIdx[tid] = (g < cnt) ? idxList[s * NATOMS + g]
                                 : idxList[s * NATOMS + tile * 16];   // pad with valid row
    }
    __syncthreads();

    // stage AEV tile: wave w loads atom row w (384 f32 -> bf16)
    {
        const float* src = aev + (size_t)atomIdx[wave] * D0;
#pragma unroll 4
        for (int e = lane; e < D0; e += 32)
            sA[wave * LDA + e] = f2bf(src[e]);
    }
    __syncthreads();

    // layer 1: [16,384] x [384,256] -> 16 waves, 12 K-steps
    {
        v8f acc = gemm_tile<12, 16>(sA, LDA, W1s, s, wave, lane);
        epilogue_bf16(acc, b1, s, D1, wave, lane, sH1, LDH1);
    }
    __syncthreads();

    // layer 2: [16,256] x [256,192] -> 12 waves, 8 K-steps
    if (wave < 12) {
        v8f acc = gemm_tile<8, 12>(sH1, LDH1, W2s, s, wave, lane);
        epilogue_bf16(acc, b2, s, D2, wave, lane, sH2, LDH2);
    }
    __syncthreads();

    // layer 3: [16,192] x [192,160] -> 10 waves, 6 K-steps (f32 out)
    if (wave < 10) {
        v8f acc = gemm_tile<6, 10>(sH2, LDH2, W3s, s, wave, lane);
        epilogue_f32(acc, b3, s, D3, wave, lane, sH3, LDH3);
    }
    __syncthreads();

    // layer 4: per-atom 160-dim dot + bias, then energy shifter.
    // wave w reduces atom w.
    {
        float sum = 0.f;
#pragma unroll
        for (int k = lane; k < D3; k += 32)
            sum += sH3[wave * LDH3 + k] * W4[s * D3 + k];
#pragma unroll
        for (int off = 16; off; off >>= 1)
            sum += __shfl_xor(sum, off, 32);
        if (lane == 0) {
            int g = tile * 16 + wave;
            if (g < cnt) {
                float coef = sum + b4[s];
                out[atomIdx[wave]] = sb0[s] + sb1[s] * coef;
            }
        }
    }
}

// ---------------------------------------------------------------------------
// launch
// ---------------------------------------------------------------------------
extern "C" void kernel_launch(void* const* d_in, const int* in_sizes, int n_in,
                              void* d_out, int out_size, void* d_ws, size_t ws_size,
                              hipStream_t stream) {
    const int*   species = (const int*)  d_in[0];
    const float* aev     = (const float*)d_in[1];
    const float* W1      = (const float*)d_in[2];
    const float* b1      = (const float*)d_in[3];
    const float* W2      = (const float*)d_in[4];
    const float* b2      = (const float*)d_in[5];
    const float* W3      = (const float*)d_in[6];
    const float* b3      = (const float*)d_in[7];
    const float* W4      = (const float*)d_in[8];
    const float* b4      = (const float*)d_in[9];
    const float* sb0     = (const float*)d_in[10];
    const float* sb1     = (const float*)d_in[11];
    float* out = (float*)d_out;

    char* ws = (char*)d_ws;
    int*      counts = (int*)     (ws + CNT_OFF);
    int*      idx    = (int*)     (ws + IDX_OFF);
    unsigned* W1s    = (unsigned*)(ws + W1S_OFF);
    unsigned* W2s    = (unsigned*)(ws + W2S_OFF);
    unsigned* W3s    = (unsigned*)(ws + W3S_OFF);

    zero_counts_kernel<<<1, 64, 0, stream>>>(counts);
    bin_kernel<<<NATOMS / 256, 256, 0, stream>>>(species, counts, idx);

    swizzle_kernel<<<768, 256, 0, stream>>>(W1, W1s, D0, D1, 16, 12);
    swizzle_kernel<<<384, 256, 0, stream>>>(W2, W2s, D1, D2, 12, 8);
    swizzle_kernel<<<240, 256, 0, stream>>>(W3, W3s, D2, D3, 10, 6);

    dim3 grid(NATOMS / 16, NSPEC);
    mlp_kernel<<<grid, 512, 0, stream>>>(aev, b1, b2, b3, b4, W4, sb0, sb1,
                                         counts, idx, W1s, W2s, W3s, out);
}
